// Listener_64390149701893
// MI455X (gfx1250) — matrix operand order repaired and checked by reading, not compiled, not run on hardware
//
#include <hip/hip_runtime.h>

#define B_   2048
#define T_   32
#define V_   32000
#define H_   1024
#define A_   1000
#define G3_  (3 * H_)
#define APAD 1008   // 63 * 16

typedef __attribute__((ext_vector_type(16))) __bf16    v16bf;
typedef __attribute__((ext_vector_type(8)))  float     v8f;
typedef __attribute__((ext_vector_type(8)))  unsigned  v8u;
typedef __attribute__((ext_vector_type(4)))  int       v4i;

typedef __attribute__((address_space(1))) void as1_void;
typedef __attribute__((address_space(3))) void as3_void;
typedef __attribute__((address_space(1))) v4i  as1_v4i;
typedef __attribute__((address_space(3))) v4i  as3_v4i;

// ---------------------------------------------------------------------------
// CDNA5 async-copy plumbing (guarded so the file always compiles)
// ---------------------------------------------------------------------------
#if __has_builtin(__builtin_amdgcn_global_load_async_to_lds_b128)
#define HAVE_ASYNC_LDS 1
#endif

__device__ __forceinline__ void async_ld_b128(const void* g, void* l) {
#if defined(HAVE_ASYNC_LDS)
  __builtin_amdgcn_global_load_async_to_lds_b128(
      (as1_v4i*)(as1_void*)(void*)g,
      (as3_v4i*)(as3_void*)(void*)l, 0, 0);
#else
  *(uint4*)l = *(const uint4*)g;   // sync fallback: global load + ds_store
#endif
}

#if defined(HAVE_ASYNC_LDS) && __has_builtin(__builtin_amdgcn_s_wait_asynccnt)
#define WAIT_ASYNC(n) __builtin_amdgcn_s_wait_asynccnt(n)
#elif defined(HAVE_ASYNC_LDS)
#define WAIT_ASYNC(n) asm volatile("s_wait_asynccnt " #n ::: "memory")
#else
#define WAIT_ASYNC(n) ((void)0)
#endif

// ---------------------------------------------------------------------------
// helpers
// ---------------------------------------------------------------------------

// pack hi16(f_lo) low half, hi16(f_hi) high half (bf16 truncation, 1 v_perm)
__device__ __forceinline__ unsigned pack_bf16(float lo, float hi) {
  return __builtin_amdgcn_perm(__builtin_bit_cast(unsigned, hi),
                               __builtin_bit_cast(unsigned, lo),
                               0x07060302u);
}

// RNE f32 -> bf16 (one-time weight conversion)
__device__ __forceinline__ unsigned short f32_to_bf16_rne(float f) {
  unsigned u = __builtin_bit_cast(unsigned, f);
  unsigned r = (u + 0x7FFFu + ((u >> 16) & 1u)) >> 16;
  return (unsigned short)r;
}

// Load 16 f32 from `row` and convert to WMMA A-operand lane layout:
//   element i (i<8) : K = koff + i          (koff = k0 + 8*(lane>>4))
//   element i (i>=8): K = koff + 16 + (i-8)
__device__ __forceinline__ v16bf load_a_cvt(const float* __restrict__ row, int koff) {
  const float4* p0 = reinterpret_cast<const float4*>(row + koff);
  const float4* p1 = reinterpret_cast<const float4*>(row + koff + 16);
  float4 a0 = p0[0], a1 = p0[1];
  float4 b0 = p1[0], b1 = p1[1];
  v8u u;
  u[0] = pack_bf16(a0.x, a0.y);  u[1] = pack_bf16(a0.z, a0.w);
  u[2] = pack_bf16(a1.x, a1.y);  u[3] = pack_bf16(a1.z, a1.w);
  u[4] = pack_bf16(b0.x, b0.y);  u[5] = pack_bf16(b0.z, b0.w);
  u[6] = pack_bf16(b1.x, b1.y);  u[7] = pack_bf16(b1.z, b1.w);
  return __builtin_bit_cast(v16bf, u);
}

__device__ __forceinline__ v8f wmma_bf16(v16bf a, v16bf b, v8f c) {
  return __builtin_amdgcn_wmma_f32_16x16x32_bf16(false, a, false, b,
                                                 (short)0, c, false, false);
}

__device__ __forceinline__ float sigmoidf_(float x) {
  return 1.0f / (1.0f + __expf(-x));
}

// ---------------------------------------------------------------------------
// prep: weights f32 -> bf16, h1 padded to APAD rows, h0 = 0, alive = 1
// ---------------------------------------------------------------------------
__global__ void prep_kernel(const float* __restrict__ wih_f,
                            const float* __restrict__ whh_f,
                            const float* __restrict__ h1_f,
                            unsigned short* __restrict__ wih_b,
                            unsigned short* __restrict__ whh_b,
                            unsigned short* __restrict__ h1_b,
                            float* __restrict__ h0,
                            int* __restrict__ alive0) {
  const int idx    = blockIdx.x * blockDim.x + threadIdx.x;
  const int stride = gridDim.x * blockDim.x;
  for (int i = idx; i < G3_ * H_; i += stride) {
    wih_b[i] = f32_to_bf16_rne(wih_f[i]);
    whh_b[i] = f32_to_bf16_rne(whh_f[i]);
  }
  for (int i = idx; i < APAD * H_; i += stride) {
    int r = i / H_;
    h1_b[i] = (r < A_) ? f32_to_bf16_rne(h1_f[i]) : (unsigned short)0;
  }
  for (int i = idx; i < B_ * H_; i += stride) h0[i] = 0.0f;
  for (int i = idx; i < B_;      i += stride) alive0[i] = 1;
}

// ---------------------------------------------------------------------------
// one fused GRU step.
// grid: (H/32, B/64), block 128 (4 waves). Each wave: 16 rows x 32 cols,
// 12 f32 WMMA accumulators (ir/iz/in/hr/hz/hn for two 16-col tiles).
// Weight tiles double-buffered in LDS via async-to-LDS DMA.
// LDS stage layout: 12 tiles [m][16 rows][32 K] bf16, m = s*6 + g*2 + ct.
// ---------------------------------------------------------------------------
#define STG_SHORTS 6144   // 12 tiles * 512 shorts = 12 KB per stage

__global__ __launch_bounds__(128) void gru_step_kernel(
    const int*            __restrict__ utt,
    const float*          __restrict__ emb_w,
    const unsigned short* __restrict__ wih,
    const unsigned short* __restrict__ whh,
    const float*          __restrict__ b_ih,
    const float*          __restrict__ b_hh,
    const float*          __restrict__ h_in,
    float*                __restrict__ h_out,
    const int*            __restrict__ alive_in,
    int*                  __restrict__ alive_out,
    int t) {
  __shared__ unsigned short lds_w[2 * STG_SHORTS];   // 24 KB double buffer

  const int tid  = threadIdx.x;
  const int lane = tid & 31;
  const int wave = tid >> 5;
  const int c0   = blockIdx.x * 32;
  const int m0   = blockIdx.y * 64 + wave * 16;

  const int half  = lane >> 4;
  const int nB    = lane & 15;
  const int kaoff = half * 8;

  // ---- async loader setup: 768 x 16B chunks per stage, 6 per thread -------
  const unsigned short* srcp[6];
  int dstoff[6];
#pragma unroll
  for (int i = 0; i < 6; ++i) {
    const int q    = tid + i * 128;
    const int m    = q >> 6;          // tile 0..11
    const int r_   = (q >> 2) & 15;   // row in tile
    const int part = q & 3;           // 16B chunk in row
    const int s    = (m >= 6);        // 0 = w_ih, 1 = w_hh
    const int rem  = m - (s ? 6 : 0);
    const int g    = rem >> 1;        // gate 0..2 (r,z,n)
    const int ct   = rem & 1;         // col tile 0..1
    const unsigned short* wb = s ? whh : wih;
    srcp[i]   = wb + (long)(g * H_ + c0 + ct * 16 + r_) * H_ + part * 8;
    dstoff[i] = m * 512 + r_ * 32 + part * 8;   // in shorts
  }
  auto issue = [&](int k0, int buf) {
#pragma unroll
    for (int i = 0; i < 6; ++i)
      async_ld_b128(srcp[i] + k0, &lds_w[buf * STG_SHORTS + dstoff[i]]);
  };

  // ---- A-operand rows: gathered embedding row + current state row ---------
  const int    arow = m0 + nB;
  const int    tok  = utt[arow * T_ + t];
  const float* xrow = emb_w + (long)tok  * H_;
  const float* hrow = h_in  + (long)arow * H_;

  v8f acc[12];
#pragma unroll
  for (int m = 0; m < 12; ++m) acc[m] = (v8f){};

  issue(0, 0);

  for (int k0 = 0, it = 0; k0 < H_; k0 += 32, ++it) {
    const bool more = (k0 + 32) < H_;
    if (more) issue(k0 + 32, (it + 1) & 1);
    if (more) { WAIT_ASYNC(6); } else { WAIT_ASYNC(0); }
    __syncthreads();

    if (more) {                              // -> global_prefetch_b8
      __builtin_prefetch(xrow + k0 + 32, 0, 1);
      __builtin_prefetch(hrow + k0 + 32, 0, 1);
    }
    v16bf ax = load_a_cvt(xrow, k0 + kaoff);
    v16bf ah = load_a_cvt(hrow, k0 + kaoff);

    const unsigned short* lb = &lds_w[(it & 1) * STG_SHORTS + nB * 32 + half * 8];
#pragma unroll
    for (int m = 0; m < 12; ++m) {
      v4i b0 = *reinterpret_cast<const v4i*>(lb + m * 512);
      v4i b1 = *reinterpret_cast<const v4i*>(lb + m * 512 + 16);
      v16bf b = __builtin_bit_cast(v16bf, (v8u){(unsigned)b0[0], (unsigned)b0[1], (unsigned)b0[2], (unsigned)b0[3], (unsigned)b1[0], (unsigned)b1[1], (unsigned)b1[2], (unsigned)b1[3]});
      acc[m] = wmma_bf16((m < 6) ? ax : ah, b, acc[m]);
    }
    __syncthreads();   // all waves done reading before this buffer is refilled
  }

  // ---- epilogue: gates + sieve for both 16-col tiles -----------------------
#pragma unroll
  for (int ct = 0; ct < 2; ++ct) {
    const int   ncol  = c0 + ct * 16 + nB;
    const float bir_s = b_ih[ncol], biz_s = b_ih[H_ + ncol], bin_s = b_ih[2 * H_ + ncol];
    const float bhr_s = b_hh[ncol], bhz_s = b_hh[H_ + ncol], bhn_s = b_hh[2 * H_ + ncol];
    const v8f air = acc[0 + ct], aiz = acc[2 + ct], ain = acc[4 + ct];
    const v8f ahr = acc[6 + ct], ahz = acc[8 + ct], ahn = acc[10 + ct];
#pragma unroll
    for (int r = 0; r < 8; ++r) {
      const int   m    = m0 + r + 8 * half;
      const float rg   = sigmoidf_((air[r] + bir_s) + (ahr[r] + bhr_s));
      const float zg   = sigmoidf_((aiz[r] + biz_s) + (ahz[r] + bhz_s));
      const float ng   = tanhf((ain[r] + bin_s) + rg * (ahn[r] + bhn_s));
      const float hold = h_in[(long)m * H_ + ncol];
      const int   al   = alive_in[m];
      const float hnew = al ? ((1.0f - zg) * ng + zg * hold) : hold;
      h_out[(long)m * H_ + ncol] = hnew;
      if (ct == 0 && blockIdx.x == 0 && nB == 0) {
        const int tk = utt[m * T_ + t];
        alive_out[m] = (al && (tk != 0)) ? 1 : 0;
      }
    }
  }
}

// ---------------------------------------------------------------------------
// logits = h @ h1_w^T + h1_b  (bf16 WMMA), padded to APAD cols
// ---------------------------------------------------------------------------
__global__ __launch_bounds__(128) void final_logits_kernel(
    const float*          __restrict__ h,
    const unsigned short* __restrict__ w1,
    const float*          __restrict__ bias,
    float*                __restrict__ logits) {
  const int lane = threadIdx.x & 31;
  const int wave = threadIdx.x >> 5;
  const int c0   = blockIdx.x * 16;
  const int m0   = blockIdx.y * 64 + wave * 16;

  const int half  = lane >> 4;
  const int kaoff = half * 8;
  const int kboff = half * 16;

  const float*          hrow = h  + (long)(m0 + (lane & 15)) * H_;
  const unsigned short* wrow = w1 + (long)(c0 + (lane & 15)) * H_;

  v8f acc = {};
  for (int k0 = 0; k0 < H_; k0 += 32) {
    v16bf a = load_a_cvt(hrow, k0 + kaoff);
    v4i b0 = *reinterpret_cast<const v4i*>(wrow + k0 + kaoff);
    v4i b1 = *reinterpret_cast<const v4i*>(wrow + k0 + kaoff + 16);
    v16bf b = __builtin_bit_cast(v16bf, (v8u){(unsigned)b0[0], (unsigned)b0[1], (unsigned)b0[2], (unsigned)b0[3], (unsigned)b1[0], (unsigned)b1[1], (unsigned)b1[2], (unsigned)b1[3]});
    acc = wmma_bf16(a, b, acc);
  }

  const int   ncol = c0 + (lane & 15);
  const float bv   = (ncol < A_) ? bias[ncol] : 0.0f;
#pragma unroll
  for (int r = 0; r < 8; ++r) {
    const int m = m0 + r + 8 * half;
    logits[(long)m * APAD + ncol] = acc[r] + bv;
  }
}

// ---------------------------------------------------------------------------
// row softmax over A_ valid columns
// ---------------------------------------------------------------------------
__global__ __launch_bounds__(128) void softmax_kernel(
    const float* __restrict__ logits, float* __restrict__ out) {
  __shared__ float red[128];
  const int    row  = blockIdx.x;
  const int    tid  = threadIdx.x;
  const float* lrow = logits + (long)row * APAD;

  float mx = -3.0e38f;
  for (int c = tid; c < A_; c += 128) mx = fmaxf(mx, lrow[c]);
  red[tid] = mx;
  __syncthreads();
  for (int s = 64; s > 0; s >>= 1) {
    if (tid < s) red[tid] = fmaxf(red[tid], red[tid + s]);
    __syncthreads();
  }
  mx = red[0];
  __syncthreads();

  float sum = 0.0f;
  for (int c = tid; c < A_; c += 128) sum += __expf(lrow[c] - mx);
  red[tid] = sum;
  __syncthreads();
  for (int s = 64; s > 0; s >>= 1) {
    if (tid < s) red[tid] += red[tid + s];
    __syncthreads();
  }
  const float inv = 1.0f / red[0];

  for (int c = tid; c < A_; c += 128)
    out[(long)row * A_ + c] = __expf(lrow[c] - mx) * inv;
}

// ---------------------------------------------------------------------------
// launcher
// ---------------------------------------------------------------------------
extern "C" void kernel_launch(void* const* d_in, const int* in_sizes, int n_in,
                              void* d_out, int out_size, void* d_ws, size_t ws_size,
                              hipStream_t stream) {
  const int*   utt   = (const int*)  d_in[0];
  // d_in[1] = global_idxes, unused by the reference output
  const float* emb_w = (const float*)d_in[2];
  const float* w_ih  = (const float*)d_in[3];
  const float* w_hh  = (const float*)d_in[4];
  const float* b_ih  = (const float*)d_in[5];
  const float* b_hh  = (const float*)d_in[6];
  const float* h1_w  = (const float*)d_in[7];
  const float* h1_b  = (const float*)d_in[8];

  char*  ws = (char*)d_ws;
  size_t o  = 0;
  auto alloc = [&](size_t bytes) -> void* {
    void* p = ws + o;
    o = (o + bytes + 255) & ~(size_t)255;
    return p;
  };
  unsigned short* wih_b  = (unsigned short*)alloc((size_t)G3_ * H_ * 2);
  unsigned short* whh_b  = (unsigned short*)alloc((size_t)G3_ * H_ * 2);
  unsigned short* h1w_b  = (unsigned short*)alloc((size_t)APAD * H_ * 2);
  float*          hA     = (float*)         alloc((size_t)B_ * H_ * 4);
  float*          hB     = (float*)         alloc((size_t)B_ * H_ * 4);
  int*            alA    = (int*)           alloc((size_t)B_ * 4);
  int*            alB    = (int*)           alloc((size_t)B_ * 4);
  float*          logits = (float*)         alloc((size_t)B_ * APAD * 4);

  prep_kernel<<<2048, 256, 0, stream>>>(w_ih, w_hh, h1_w,
                                        wih_b, whh_b, h1w_b, hA, alA);

  for (int t = 0; t < T_; ++t) {
    const float* hin  = (t & 1) ? hB  : hA;
    float*       hout = (t & 1) ? hA  : hB;
    const int*   ain  = (t & 1) ? alB : alA;
    int*         aout = (t & 1) ? alA : alB;
    gru_step_kernel<<<dim3(H_ / 32, B_ / 64), 128, 0, stream>>>(
        utt, emb_w, wih_b, whh_b, b_ih, b_hh, hin, hout, ain, aout, t);
  }

  // T_=32 steps -> final state back in hA
  final_logits_kernel<<<dim3(APAD / 16, B_ / 64), 128, 0, stream>>>(
      hA, h1w_b, h1_b, logits);

  softmax_kernel<<<B_, 128, 0, stream>>>(logits, (float*)d_out);
}
